// LSQBlock_84885733638932
// MI455X (gfx1250) — matrix operand order repaired and checked by reading, not compiled, not run on hardware
//
#include <hip/hip_runtime.h>
#include <hip/hip_bf16.h>
#include <math.h>

// ---------------------------------------------------------------------------
// ViT encoder block for MI455X (gfx1250, wave32, WMMA + TDM).
// B=8, N=1024, C=768, HEADS=12, HD=64, MLP=3072. fp32 I/O, bf16 WMMA compute.
// ---------------------------------------------------------------------------

typedef __bf16 bf16;
typedef __attribute__((ext_vector_type(8)))  __bf16 v8bf;
typedef __attribute__((ext_vector_type(16))) __bf16 v16bf;
typedef __attribute__((ext_vector_type(8)))  float  v8f;
typedef __attribute__((ext_vector_type(4)))  unsigned int u32x4;
typedef __attribute__((ext_vector_type(8)))  int i32x8;
typedef __attribute__((ext_vector_type(4)))  int i32x4;

#define BB   8
#define NN   1024
#define CC   768
#define NH   12
#define HD   64
#define C3   2304
#define MLPD 3072
#define MTOT (BB * NN)          // 8192 rows

#if defined(__has_builtin)
#if __has_builtin(__builtin_amdgcn_tensor_load_to_lds) && \
    __has_builtin(__builtin_amdgcn_s_wait_tensorcnt)
#define USE_TDM 1
#endif
#endif
#ifndef USE_TDM
#define USE_TDM 0
#endif

__device__ __forceinline__ v8f wmma_bf16(v16bf a, v16bf b, v8f c) {
    return __builtin_amdgcn_wmma_f32_16x16x32_bf16(
        /*neg_a=*/false, a, /*neg_b=*/false, b,
        /*c_mod=*/(short)0, c, /*reuse_a=*/false, /*reuse_b=*/false);
}

// Build a 16-wide bf16 fragment from two 16-byte LDS chunks (-> 2x ds_load_b128)
__device__ __forceinline__ v16bf frag_from(const bf16* p0, const bf16* p1) {
    v8bf lo = *(const v8bf*)p0;
    v8bf hi = *(const v8bf*)p1;
    v16bf r;
#pragma unroll
    for (int i = 0; i < 8; ++i) { r[i] = lo[i]; r[i + 8] = hi[i]; }
    return r;
}

__device__ __forceinline__ v8f v8f_zero() {
    v8f z = {0.f, 0.f, 0.f, 0.f, 0.f, 0.f, 0.f, 0.f};
    return z;
}

#if USE_TDM
// LDS byte offset of a __shared__ object (low 32 bits of the flat address)
__device__ __forceinline__ unsigned lds_off(const void* p) {
    return (unsigned)(size_t)p;
}

// Issue one TDM 2D-tile load: tile_rows x 32 bf16 elements from a row-major
// tensor with row stride `stride_elems`, into LDS rows padded to 40 elements
// (64B data + 16B pad per row -> pad_interval=16 DW, pad_amount=4 DW).
// This toolchain's builtin takes 6 args: (g0, g1, g2, g3, g_extra, cpol).
__device__ __forceinline__ void tdm_load_tile(const bf16* gsrc, unsigned ldsaddr,
                                              int tile_rows, int stride_elems) {
    unsigned long long ga = (unsigned long long)(size_t)gsrc;
    u32x4 g0;
    g0[0] = 1u;                                           // count=1, user desc
    g0[1] = ldsaddr;                                      // lds_addr
    g0[2] = (unsigned)ga;                                 // global_addr[31:0]
    g0[3] = (unsigned)((ga >> 32) & 0x01FFFFFFu)          // global_addr[56:32]
            | 0x80000000u;                                // type=2 ("image")
    i32x8 g1;
    g1[0] = (1 << 16)       // data_size = 1 (2 bytes)
          | (1 << 20)       // pad_enable
          | (3 << 22)       // pad_interval: 16 DWORDs (64B)
          | (3 << 25);      // pad_amount:   4 DWORDs (16B)
    g1[1] = (stride_elems & 0xFFFF) << 16;                // tensor_dim0[15:0]
    g1[2] = ((unsigned)stride_elems >> 16)                // tensor_dim0[31:16]
          | (MTOT << 16);                                 // tensor_dim1[15:0]
    g1[3] = (32 << 16);                                   // tile_dim0 = 32
    g1[4] = tile_rows;                                    // tile_dim1
    g1[5] = stride_elems;                                 // tensor_dim0_stride lo
    g1[6] = 0;
    g1[7] = 0;
    i32x4 z4 = {0, 0, 0, 0};
    i32x8 z8 = {0, 0, 0, 0, 0, 0, 0, 0};
    __builtin_amdgcn_tensor_load_to_lds(g0, g1, z4, z4, z8, 0);
}
#endif

// ---------------------------------------------------------------------------
// fp32 -> bf16 conversion (weights), grid-stride
// ---------------------------------------------------------------------------
__global__ __launch_bounds__(256) void cvt_kernel(const float* __restrict__ in,
                                                  bf16* __restrict__ out, int n) {
    int i = blockIdx.x * blockDim.x + threadIdx.x;
    int stride = gridDim.x * blockDim.x;
    for (; i < n; i += stride) out[i] = (bf16)in[i];
}

// ---------------------------------------------------------------------------
// LayerNorm over C=768, one block (256 thr) per row; bf16 output
// ---------------------------------------------------------------------------
__global__ __launch_bounds__(256) void ln_kernel(const float* __restrict__ x,
                                                 const float* __restrict__ g,
                                                 const float* __restrict__ b,
                                                 bf16* __restrict__ out) {
    __shared__ float rs[256], rss[256];
    const int row = blockIdx.x, tid = threadIdx.x;
    const float* xr = x + (size_t)row * CC;
    float v[3], s = 0.f, ss = 0.f;
#pragma unroll
    for (int j = 0; j < 3; ++j) {
        v[j] = xr[tid + j * 256];
        s += v[j];
        ss += v[j] * v[j];
    }
    rs[tid] = s; rss[tid] = ss;
    __syncthreads();
    for (int off = 128; off > 0; off >>= 1) {
        if (tid < off) { rs[tid] += rs[tid + off]; rss[tid] += rss[tid + off]; }
        __syncthreads();
    }
    const float mu   = rs[0] * (1.f / CC);
    const float var  = rss[0] * (1.f / CC) - mu * mu;
    const float rinv = rsqrtf(var + 1e-6f);
    bf16* orow = out + (size_t)row * CC;
#pragma unroll
    for (int j = 0; j < 3; ++j) {
        int c = tid + j * 256;
        orow[c] = (bf16)((v[j] - mu) * rinv * g[c] + b[c]);
    }
}

// ---------------------------------------------------------------------------
// GEMM: Out[M,D] = act(A[M,K] @ W[D,K]^T + bias) (+ residual)
// A, W bf16 row-major. Block tile 64x128, 8 waves (2x4), 2x2 WMMA tiles/wave.
// Tile staging: TDM (tensor_load_to_lds) with double buffering when available,
// else synchronous b128 global->LDS copies.
// flags: bit0 = exact GELU, bit1 = bf16 output (else fp32)
// ---------------------------------------------------------------------------
__global__ __launch_bounds__(256) void gemm_kernel(const bf16* __restrict__ Ag,
                                                   const bf16* __restrict__ Wg,
                                                   const float* __restrict__ bias,
                                                   const float* __restrict__ residual,
                                                   void* __restrict__ outp,
                                                   int M, int K, int D, int flags) {
    const int tid  = threadIdx.x;
    const int lane = tid & 31;
    const int w    = tid >> 5;
    const int half = lane >> 4;
    const int l16  = lane & 15;
    const int waveM = w >> 2;        // 0..1  (32 rows each)
    const int waveN = w & 3;         // 0..3  (32 cols each)
    const int m0 = blockIdx.y * 64;
    const int n0 = blockIdx.x * 128;

    v8f acc[2][2];
#pragma unroll
    for (int tm = 0; tm < 2; ++tm)
#pragma unroll
        for (int tn = 0; tn < 2; ++tn) acc[tm][tn] = v8f_zero();

    const int nkt = K >> 5;

#if USE_TDM
    __shared__ __align__(16) bf16 As[2][64][40];
    __shared__ __align__(16) bf16 Bs[2][128][40];

    if (w == 0) {   // prologue: DMA tile 0 into buffer 0
        tdm_load_tile(Ag + (size_t)m0 * K, lds_off(&As[0][0][0]), 64, K);
        tdm_load_tile(Wg + (size_t)n0 * K, lds_off(&Bs[0][0][0]), 128, K);
    }
    for (int kt = 0; kt < nkt; ++kt) {
        const int cur = kt & 1;
        __syncthreads();               // all waves done reading buf[1-cur]
        if (kt + 1 < nkt) {
            if (w == 0) {              // overlap next tile's DMA with compute
                tdm_load_tile(Ag + (size_t)m0 * K + (kt + 1) * 32,
                              lds_off(&As[1 - cur][0][0]), 64, K);
                tdm_load_tile(Wg + (size_t)n0 * K + (kt + 1) * 32,
                              lds_off(&Bs[1 - cur][0][0]), 128, K);
            }
            __builtin_amdgcn_s_wait_tensorcnt(2);   // tile kt landed
        } else {
            __builtin_amdgcn_s_wait_tensorcnt(0);
        }
        __syncthreads();               // broadcast tile-kt completion

        v16bf af[2], bfv[2];
#pragma unroll
        for (int tm = 0; tm < 2; ++tm) {
            const bf16* base = &As[cur][waveM * 32 + tm * 16 + l16][0];
            af[tm] = frag_from(base + 8 * half, base + 16 + 8 * half);
        }
#pragma unroll
        for (int tn = 0; tn < 2; ++tn) {
            const bf16* base = &Bs[cur][waveN * 32 + tn * 16 + l16][0];
            bfv[tn] = frag_from(base + 8 * half, base + 16 + 8 * half);
        }
#pragma unroll
        for (int tm = 0; tm < 2; ++tm)
#pragma unroll
            for (int tn = 0; tn < 2; ++tn)
                acc[tm][tn] = wmma_bf16(af[tm], bfv[tn], acc[tm][tn]);
    }
#else
    __shared__ __align__(16) bf16 As[64][40];
    __shared__ __align__(16) bf16 Bs[128][40];

    for (int kt = 0; kt < nkt; ++kt) {
        __syncthreads();
        {   // stage A tile 64x32 (one uint4 per thread)
            int r = tid >> 2, c = (tid & 3) * 8;
            const uint4* src = (const uint4*)(Ag + (size_t)(m0 + r) * K + kt * 32 + c);
            *(uint4*)&As[r][c] = src[0];
        }
#pragma unroll
        for (int i = 0; i < 2; ++i) {  // stage B tile 128x32 (two uint4 per thread)
            int ch = tid + i * 256;
            int r = ch >> 2, c = (ch & 3) * 8;
            const uint4* src = (const uint4*)(Wg + (size_t)(n0 + r) * K + kt * 32 + c);
            *(uint4*)&Bs[r][c] = src[0];
        }
        __syncthreads();

        v16bf af[2], bfv[2];
#pragma unroll
        for (int tm = 0; tm < 2; ++tm) {
            const bf16* base = &As[waveM * 32 + tm * 16 + l16][0];
            af[tm] = frag_from(base + 8 * half, base + 16 + 8 * half);
        }
#pragma unroll
        for (int tn = 0; tn < 2; ++tn) {
            const bf16* base = &Bs[waveN * 32 + tn * 16 + l16][0];
            bfv[tn] = frag_from(base + 8 * half, base + 16 + 8 * half);
        }
#pragma unroll
        for (int tm = 0; tm < 2; ++tm)
#pragma unroll
            for (int tn = 0; tn < 2; ++tn)
                acc[tm][tn] = wmma_bf16(af[tm], bfv[tn], acc[tm][tn]);
    }
#endif

    // epilogue
#pragma unroll
    for (int tm = 0; tm < 2; ++tm)
#pragma unroll
        for (int tn = 0; tn < 2; ++tn)
#pragma unroll
            for (int i = 0; i < 8; ++i) {
                int row = m0 + waveM * 32 + tm * 16 + i + 8 * half;
                int col = n0 + waveN * 32 + tn * 16 + l16;
                float v = acc[tm][tn][i] + bias[col];
                if (residual) v += residual[(size_t)row * D + col];
                if (flags & 1) v = 0.5f * v * (1.0f + erff(v * 0.70710678118654752f));
                if (flags & 2) ((bf16*)outp)[(size_t)row * D + col] = (bf16)v;
                else           ((float*)outp)[(size_t)row * D + col] = v;
            }
}

// ---------------------------------------------------------------------------
// Fused flash attention. Grid: (N/64, B*NH). One block = 64 query rows of one
// head. Streams 64-key tiles; online softmax; S = Q K^T and O += P V via WMMA.
// qkv layout: [B, N, 3C] bf16 with q at d = h*64, k at C + h*64, v at 2C + h*64.
// Output o bf16 [B, N, C].
// ---------------------------------------------------------------------------
__global__ __launch_bounds__(256) void attn_kernel(const bf16* __restrict__ qkv,
                                                   bf16* __restrict__ o) {
    __shared__ __align__(16) bf16 Qs[64][72];
    __shared__ __align__(16) bf16 Ks[64][72];
    __shared__ __align__(16) bf16 Vt[64][72];   // transposed: Vt[hd][key]
    __shared__ __align__(16) bf16 Ps[64][72];
    __shared__ float rowMax[64], rowSum[64], alphaS[64], mnewS[64];
    __shared__ float wred[2][64];

    const int tid  = threadIdx.x;
    const int lane = tid & 31;
    const int w    = tid >> 5;
    const int half = lane >> 4;
    const int l16  = lane & 15;
    const int wm = w >> 1;           // 0..3: 16-row strip of the 64 q rows
    const int wn = w & 1;            // 0..1: 32-col strip (keys / hd)
    const int q0 = blockIdx.x * 64;
    const int bh = blockIdx.y;
    const int b  = bh / NH, hh = bh % NH;
    const bf16* qbase = qkv + (size_t)b * NN * C3 + hh * HD;

    if (tid < 64) { rowMax[tid] = -3.0e38f; rowSum[tid] = 0.f; }

    {   // stage Q tile 64x64 (two uint4 per thread)
        int r = tid >> 2, c = (tid & 3) * 16;
        const uint4* src = (const uint4*)(qbase + (size_t)(q0 + r) * C3 + c);
        *(uint4*)&Qs[r][c]     = src[0];
        *(uint4*)&Qs[r][c + 8] = src[1];
    }

    v8f oacc[2];
    oacc[0] = v8f_zero(); oacc[1] = v8f_zero();

    for (int kt = 0; kt < NN / 64; ++kt) {
        const int k0 = kt * 64;
        __syncthreads();               // prior P@V done; safe to refill K/V
        {   // stage K tile (row major) and V tile (transposed into Vt)
            int r = tid >> 2, c = (tid & 3) * 16;
            const uint4* ks = (const uint4*)(qbase + CC + (size_t)(k0 + r) * C3 + c);
            *(uint4*)&Ks[r][c]     = ks[0];
            *(uint4*)&Ks[r][c + 8] = ks[1];
            const uint4* vs = (const uint4*)(qbase + 2 * CC + (size_t)(k0 + r) * C3 + c);
            union { uint4 u; bf16 h[8]; } t0, t1;
            t0.u = vs[0]; t1.u = vs[1];
#pragma unroll
            for (int j = 0; j < 8; ++j) {
                Vt[c + j][r]     = t0.h[j];
                Vt[c + 8 + j][r] = t1.h[j];
            }
            if (kt + 1 < NN / 64) {    // pull next K/V tiles toward L2
                __builtin_prefetch(qbase + CC + (size_t)(k0 + 64 + r) * C3 + c, 0, 0);
                __builtin_prefetch(qbase + 2 * CC + (size_t)(k0 + 64 + r) * C3 + c, 0, 0);
            }
        }
        __syncthreads();

        // S = scale * Q K^T  (wave: rows wm*16.., keys wn*32.. as 2 tiles)
        v8f sacc[2];
        sacc[0] = v8f_zero(); sacc[1] = v8f_zero();
#pragma unroll
        for (int ks = 0; ks < 2; ++ks) {
            const bf16* abase = &Qs[wm * 16 + l16][ks * 32];
            v16bf aq = frag_from(abase + 8 * half, abase + 16 + 8 * half);
#pragma unroll
            for (int tn = 0; tn < 2; ++tn) {
                const bf16* bbase = &Ks[wn * 32 + tn * 16 + l16][ks * 32];
                v16bf bk = frag_from(bbase + 8 * half, bbase + 16 + 8 * half);
                sacc[tn] = wmma_bf16(aq, bk, sacc[tn]);
            }
        }
        // scale + per-row max across the wave's 32 keys
#pragma unroll
        for (int i = 0; i < 8; ++i) {
            sacc[0][i] *= 0.125f;
            sacc[1][i] *= 0.125f;
            float m = fmaxf(sacc[0][i], sacc[1][i]);
            m = fmaxf(m, __shfl_xor(m, 1, 32));
            m = fmaxf(m, __shfl_xor(m, 2, 32));
            m = fmaxf(m, __shfl_xor(m, 4, 32));
            m = fmaxf(m, __shfl_xor(m, 8, 32));
            if (l16 == 0) wred[wn][wm * 16 + i + 8 * half] = m;
        }
        __syncthreads();
        if (tid < 64) {
            float mo = rowMax[tid];
            float mn = fmaxf(mo, fmaxf(wred[0][tid], wred[1][tid]));
            rowMax[tid] = mn;
            alphaS[tid] = __expf(mo - mn);
            mnewS[tid]  = mn;
        }
        __syncthreads();

        // P = exp(S - mnew); stage Ps; partial row sums; rescale O
#pragma unroll
        for (int i = 0; i < 8; ++i) {
            int row = wm * 16 + i + 8 * half;
            float mn = mnewS[row];
            float p0 = __expf(sacc[0][i] - mn);
            float p1 = __expf(sacc[1][i] - mn);
            Ps[row][wn * 32 + l16]      = (bf16)p0;
            Ps[row][wn * 32 + 16 + l16] = (bf16)p1;
            float s = p0 + p1;
            s += __shfl_xor(s, 1, 32);
            s += __shfl_xor(s, 2, 32);
            s += __shfl_xor(s, 4, 32);
            s += __shfl_xor(s, 8, 32);
            if (l16 == 0) wred[wn][row] = s;
            float a = alphaS[row];
            oacc[0][i] *= a;
            oacc[1][i] *= a;
        }
        __syncthreads();               // Ps + wred visible
        if (tid < 64)
            rowSum[tid] = rowSum[tid] * alphaS[tid] + wred[0][tid] + wred[1][tid];

        // O += P V   (wave: rows wm*16.., hd cols wn*32.. as 2 tiles)
#pragma unroll
        for (int ks = 0; ks < 2; ++ks) {
            const bf16* abase = &Ps[wm * 16 + l16][ks * 32];
            v16bf ap = frag_from(abase + 8 * half, abase + 16 + 8 * half);
#pragma unroll
            for (int tn = 0; tn < 2; ++tn) {
                const bf16* bbase = &Vt[wn * 32 + tn * 16 + l16][ks * 32];
                v16bf bv = frag_from(bbase + 8 * half, bbase + 16 + 8 * half);
                oacc[tn] = wmma_bf16(ap, bv, oacc[tn]);
            }
        }
    }
    __syncthreads();

    // normalize by rowSum and write o (bf16)
#pragma unroll
    for (int tn = 0; tn < 2; ++tn)
#pragma unroll
        for (int i = 0; i < 8; ++i) {
            int row = wm * 16 + i + 8 * half;
            int col = wn * 32 + tn * 16 + l16;
            float v = oacc[tn][i] / rowSum[row];
            o[((size_t)b * NN + q0 + row) * CC + hh * HD + col] = (bf16)v;
        }
}

// ---------------------------------------------------------------------------
// Host launch: convert weights -> bf16, then
// LN1 -> QKV gemm -> flash attn -> proj gemm(+x) -> LN2 -> FC1 gemm(GELU) ->
// FC2 gemm(+x1) -> d_out
// ---------------------------------------------------------------------------
extern "C" void kernel_launch(void* const* d_in, const int* in_sizes, int n_in,
                              void* d_out, int out_size, void* d_ws, size_t ws_size,
                              hipStream_t stream) {
    (void)in_sizes; (void)n_in; (void)out_size; (void)ws_size;
    const float* x      = (const float*)d_in[0];
    const float* ln1_w  = (const float*)d_in[3];
    const float* ln1_b  = (const float*)d_in[4];
    const float* qkv_w  = (const float*)d_in[5];
    const float* qkv_b  = (const float*)d_in[6];
    const float* proj_w = (const float*)d_in[7];
    const float* proj_b = (const float*)d_in[8];
    const float* ln2_w  = (const float*)d_in[9];
    const float* ln2_b  = (const float*)d_in[10];
    const float* fc1_w  = (const float*)d_in[11];
    const float* fc1_b  = (const float*)d_in[12];
    const float* fc2_w  = (const float*)d_in[13];
    const float* fc2_b  = (const float*)d_in[14];
    float* out = (float*)d_out;

    // workspace layout (all 16B aligned)
    bf16* wq    = (bf16*)d_ws;
    bf16* wproj = wq    + (size_t)C3 * CC;
    bf16* wfc1  = wproj + (size_t)CC * CC;
    bf16* wfc2  = wfc1  + (size_t)MLPD * CC;
    bf16* hbf   = wfc2  + (size_t)CC * MLPD;          // LN output [M, C]
    bf16* qkvbf = hbf   + (size_t)MTOT * CC;          // [M, 3C]
    bf16* obf   = qkvbf + (size_t)MTOT * C3;          // [M, C]
    float* x1   = (float*)(obf + (size_t)MTOT * CC);  // [M, C] fp32
    bf16* gbf   = qkvbf;  // FC1 output [M, MLP] aliases qkv+o (free by then)

    // 1. weights -> bf16
    cvt_kernel<<<dim3(1024), dim3(256), 0, stream>>>(qkv_w,  wq,    C3 * CC);
    cvt_kernel<<<dim3(512),  dim3(256), 0, stream>>>(proj_w, wproj, CC * CC);
    cvt_kernel<<<dim3(1024), dim3(256), 0, stream>>>(fc1_w,  wfc1,  MLPD * CC);
    cvt_kernel<<<dim3(1024), dim3(256), 0, stream>>>(fc2_w,  wfc2,  CC * MLPD);

    // 2. LN1
    ln_kernel<<<dim3(MTOT), dim3(256), 0, stream>>>(x, ln1_w, ln1_b, hbf);
    // 3. QKV gemm -> bf16
    gemm_kernel<<<dim3(C3 / 128, MTOT / 64), dim3(256), 0, stream>>>(
        hbf, wq, qkv_b, nullptr, qkvbf, MTOT, CC, C3, /*flags=*/2);
    // 4. fused attention -> obf
    attn_kernel<<<dim3(NN / 64, BB * NH), dim3(256), 0, stream>>>(qkvbf, obf);
    // 5. proj gemm + residual x -> x1 (fp32)
    gemm_kernel<<<dim3(CC / 128, MTOT / 64), dim3(256), 0, stream>>>(
        obf, wproj, proj_b, x, x1, MTOT, CC, CC, /*flags=*/0);
    // 6. LN2
    ln_kernel<<<dim3(MTOT), dim3(256), 0, stream>>>(x1, ln2_w, ln2_b, hbf);
    // 7. FC1 gemm + GELU -> bf16
    gemm_kernel<<<dim3(MLPD / 128, MTOT / 64), dim3(256), 0, stream>>>(
        hbf, wfc1, fc1_b, nullptr, gbf, MTOT, CC, MLPD, /*flags=*/1 | 2);
    // 8. FC2 gemm + residual x1 -> d_out (fp32)
    gemm_kernel<<<dim3(CC / 128, MTOT / 64), dim3(256), 0, stream>>>(
        gbf, wfc2, fc2_b, x1, out, MTOT, MLPD, CC, /*flags=*/0);
}
